// LinearVectorQuantizer_24094766531139
// MI455X (gfx1250) — matrix operand order repaired and checked by reading, not compiled, not run on hardware
//
#include <hip/hip_runtime.h>

typedef __attribute__((ext_vector_type(16))) __bf16 v16bf;
typedef __attribute__((ext_vector_type(8)))  float  v8f;
typedef __attribute__((ext_vector_type(4)))  unsigned int u32x4;
typedef __attribute__((ext_vector_type(8)))  int i32x8;
typedef __attribute__((ext_vector_type(4)))  int i32x4;

#define D_DIM 512
#define TILE_N 32                 // codebook rows staged per iteration
#define WAVES_PER_BLOCK 8
#define ROWS_PER_WAVE 16
#define ROWS_PER_BLOCK (WAVES_PER_BLOCK * ROWS_PER_WAVE)
#define CB_LDS_PITCH 520          // ushorts/row: 512 + 8 pad -> bank stride 4, conflict-free b128

// ---- Tensor Data Mover availability (device pass only; host pass sees 0 -> fallback parse) ----
#if defined(__has_builtin)
#if __has_builtin(__builtin_amdgcn_tensor_load_to_lds)
#define USE_TDM 1
#else
#define USE_TDM 0
#endif
#else
#define USE_TDM 0
#endif

__device__ __forceinline__ unsigned short f2bf(float f) {
  unsigned u = __float_as_uint(f);
  u += 0x7FFFu + ((u >> 16) & 1u);   // round-to-nearest-even
  return (unsigned short)(u >> 16);
}

__device__ __forceinline__ void wait_tensorcnt0() {
#if defined(__has_builtin)
#if __has_builtin(__builtin_amdgcn_s_wait_tensorcnt)
  __builtin_amdgcn_s_wait_tensorcnt(0);
#else
  asm volatile("s_wait_tensorcnt 0x0" ::: "memory");
#endif
#else
  asm volatile("s_wait_tensorcnt 0x0" ::: "memory");
#endif
}

#if USE_TDM
// Issue one TDM transfer: TILE_N x 512 bf16 tile (row-major, stride 512) -> LDS with
// 16B padding after every 1024B row (pad_interval=7 -> 256 DWORDs, pad_amount=3 -> 4 DWORDs),
// producing the 520-ushort LDS pitch directly. 2-D descriptor: trailing groups zero.
// This toolchain exposes the 6-arg builtin: (u32x4, i32x8, i32x4, i32x4, i32x8, i32 cpol).
__device__ __forceinline__ void tdm_issue_tile(const unsigned short* gsrc, unsigned ldsByteAddr) {
  unsigned long long ga = (unsigned long long)(uintptr_t)gsrc;
  u32x4 g0;
  g0.x = 1u;                                                  // count=1, user descriptor
  g0.y = ldsByteAddr;                                         // lds_addr [63:32]
  g0.z = (unsigned)(ga & 0xFFFFFFFFu);                        // global_addr [95:64]
  g0.w = (unsigned)((ga >> 32) & 0x01FFFFFFu) | (2u << 30);   // global_addr hi + type=2
  i32x8 g1;
  g1[0] = (int)((1u << 16) |        // data_size = 2 bytes
                (1u << 20) |        // pad_enable
                (7u << 22) |        // pad_interval: 256 DWORDs (= one 1024B row)
                (3u << 25));        // pad_amount: 4 DWORDs (16B -> pitch 520 ushorts)
  g1[1] = (int)(512u << 16);        // tensor_dim0 = 512 (low 16 in d1[31:16])
  g1[2] = (int)((unsigned)TILE_N << 16);  // tensor_dim0 hi=0, tensor_dim1 = TILE_N
  g1[3] = (int)(512u << 16);        // tensor_dim1 hi=0, tile_dim0 = 512
  g1[4] = (int)TILE_N;              // tile_dim1 = TILE_N, tile_dim2 = 0
  g1[5] = 512;                      // tensor_dim0_stride = 512 elements
  g1[6] = 0;                        // stride hi, tensor_dim1_stride lo
  g1[7] = 0;
  i32x4 z4 = {0, 0, 0, 0};
  i32x8 z8 = {0, 0, 0, 0, 0, 0, 0, 0};
  __builtin_amdgcn_tensor_load_to_lds(g0, g1, z4, z4, z8, 0);
}
#endif

// Kernel 1: exact f32 ||codebook_k||^2 (one wave per row) + optional bf16 pre-conversion.
__global__ __launch_bounds__(256) void vq_prep_kernel(const float* __restrict__ cb,
                                                      float* __restrict__ cbsq,
                                                      unsigned short* __restrict__ cbB, int K) {
  int wave = threadIdx.x >> 5;
  int lane = threadIdx.x & 31;
  int row  = blockIdx.x * WAVES_PER_BLOCK + wave;
  if (row >= K) return;
  const float* p = cb + (size_t)row * D_DIM;
  float s = 0.f;
#pragma unroll
  for (int j = 0; j < 4; ++j) {
    int base = (lane + 32 * j) * 4;
    float4 v = *(const float4*)(p + base);
    s += v.x * v.x + v.y * v.y + v.z * v.z + v.w * v.w;
    if (cbB) {
      ushort4 b;
      b.x = f2bf(v.x); b.y = f2bf(v.y); b.z = f2bf(v.z); b.w = f2bf(v.w);
      *(ushort4*)(cbB + (size_t)row * D_DIM + base) = b;
    }
  }
  for (int m = 16; m >= 1; m >>= 1) s += __shfl_xor(s, m, 32);
  if (lane == 0) cbsq[row] = s;
}

// Kernel 2: fused bf16-WMMA GEMM + argmin + exact f32 gather.
// PRE=true: codebook already bf16 in cbB; stage via TDM (double-buffered) or plain copy.
// PRE=false: stage from f32 codebook with in-loop conversion.
template <bool PRE>
__global__ __launch_bounds__(256) void vq_main_kernel(
    const float* __restrict__ x, const float* __restrict__ cb,
    const unsigned short* __restrict__ cbB, const float* __restrict__ cbsq,
    float* __restrict__ out_e, float* __restrict__ out_ids, int K) {
  constexpr int NBUF = (PRE && USE_TDM) ? 2 : 1;
  __shared__ unsigned short cbTile[NBUF * TILE_N * CB_LDS_PITCH];
  __shared__ int idsBuf[WAVES_PER_BLOCK * ROWS_PER_WAVE];

  const int tid  = threadIdx.x;
  const int wave = tid >> 5;
  const int lane = tid & 31;
  const int l16  = lane & 15;
  const int hi   = lane >> 4;
  const int rowBase = (blockIdx.x * WAVES_PER_BLOCK + wave) * ROWS_PER_WAVE;

  // ---- Resident A fragments: this wave's 16x512 x-tile as bf16 (16 frags x 8 VGPRs) ----
  union AF { v16bf v; unsigned short s[16]; };
  AF afrag[16];
  {
    const float* xr = x + (size_t)(rowBase + l16) * D_DIM;
#pragma unroll
    for (int f = 0; f < 16; ++f) {
      const int d0 = 32 * f + 16 * hi;
      const int d1 = 32 * f + 16 * hi + 8;
#pragma unroll
      for (int k = 0; k < 8; ++k) afrag[f].s[k]     = f2bf(xr[d0 + k]);
#pragma unroll
      for (int k = 0; k < 8; ++k) afrag[f].s[8 + k] = f2bf(xr[d1 + k]);
    }
  }

  float minv[8];
  int   mini[8];
#pragma unroll
  for (int v = 0; v < 8; ++v) { minv[v] = 3.4e38f; mini[v] = 0; }

#if USE_TDM
  if constexpr (PRE) {
    if (wave == 0)   // TDM prologue: fill buffer 0 with the first tile
      tdm_issue_tile(cbB, (unsigned)(uintptr_t)&cbTile[0]);
  }
#endif

  for (int nt = 0; nt < K; nt += TILE_N) {
    int cur = 0;
    if constexpr (PRE) {
#if USE_TDM
      cur = (nt / TILE_N) & 1;
      if (wave == 0) wait_tensorcnt0();       // tile `cur` landed in LDS
      __syncthreads();
      if (wave == 0 && nt + TILE_N < K)       // overlap: DMA next tile into other buffer
        tdm_issue_tile(cbB + (size_t)(nt + TILE_N) * D_DIM,
                       (unsigned)(uintptr_t)&cbTile[(cur ^ 1) * TILE_N * CB_LDS_PITCH]);
#else
      __syncthreads();
      // synchronous bf16 tile copy: 32KB / 256 threads = 8 x b128 per thread
#pragma unroll
      for (int j = 0; j < 8; ++j) {
        int c = tid + 256 * j;
        int r = c >> 6, col16 = c & 63;       // 64 x 16B chunks per 1024B row
        const uint4* src = (const uint4*)(cbB + (size_t)(nt + r) * D_DIM) + col16;
        *((uint4*)&cbTile[r * CB_LDS_PITCH] + col16) = *src;
      }
      __syncthreads();
#endif
    } else {
      __syncthreads();
      // f32 -> bf16 staging: 32x512 elems, 16 float4 per thread
#pragma unroll 4
      for (int j = 0; j < 16; ++j) {
        int c4 = tid + 256 * j;
        int r = c4 >> 7, col4 = c4 & 127;
        float4 v = *(const float4*)(cb + (size_t)(nt + r) * D_DIM + col4 * 4);
        ushort4 b;
        b.x = f2bf(v.x); b.y = f2bf(v.y); b.z = f2bf(v.z); b.w = f2bf(v.w);
        *(ushort4*)&cbTile[r * CB_LDS_PITCH + col4 * 4] = b;
      }
      __syncthreads();
    }

    const unsigned short* tile = &cbTile[cur * TILE_N * CB_LDS_PITCH];

    // Two 16-column sub-tiles; 16 WMMAs each, B loads prefetched 2 iterations ahead.
#pragma unroll
    for (int s = 0; s < 2; ++s) {
      const unsigned short* base = tile + (s * 16 + l16) * CB_LDS_PITCH + 16 * hi;
      union BB { v16bf v; uint4 q[2]; } bb[4];
#pragma unroll
      for (int p = 0; p < 2; ++p) {
        const uint4* q = (const uint4*)(base + 32 * p);
        bb[p].q[0] = q[0]; bb[p].q[1] = q[1];
      }
      v8f acc0 = {};
      v8f acc1 = {};
#pragma unroll
      for (int f = 0; f < 16; ++f) {
        if (f < 14) {
          const uint4* q = (const uint4*)(base + 32 * (f + 2));
          bb[(f + 2) & 3].q[0] = q[0];
          bb[(f + 2) & 3].q[1] = q[1];
        }
        if (f & 1)
          acc1 = __builtin_amdgcn_wmma_f32_16x16x32_bf16(false, afrag[f].v, false, bb[f & 3].v,
                                                         (short)0, acc1, false, false);
        else
          acc0 = __builtin_amdgcn_wmma_f32_16x16x32_bf16(false, afrag[f].v, false, bb[f & 3].v,
                                                         (short)0, acc0, false, false);
      }
      const int   col = nt + s * 16 + l16;
      const float csq = cbsq[col];
#pragma unroll
      for (int v = 0; v < 8; ++v) {
        float dist = csq - 2.0f * (acc0[v] + acc1[v]);
        if (dist < minv[v]) { minv[v] = dist; mini[v] = col; }  // strict < keeps first min
      }
    }
    __syncthreads();   // compute done before this buffer may be overwritten
  }

  // ---- Argmin reduction across the 16 lanes of each half-wave ----
#pragma unroll
  for (int v = 0; v < 8; ++v) {
    float val = minv[v];
    int   idx = mini[v];
#pragma unroll
    for (int m = 1; m < 16; m <<= 1) {
      float ov = __shfl_xor(val, m, 32);
      int   oi = __shfl_xor(idx, m, 32);
      if (ov < val || (ov == val && oi < idx)) { val = ov; idx = oi; }
    }
    minv[v] = val; mini[v] = idx;
  }

  if (l16 == 0) {
#pragma unroll
    for (int v = 0; v < 8; ++v) {
      int r = v + 8 * hi;
      idsBuf[wave * ROWS_PER_WAVE + r] = mini[v];
      out_ids[rowBase + r] = (float)mini[v];
    }
  }
  __syncthreads();

  // ---- Gather winning rows from the exact f32 codebook ----
  for (int r = 0; r < ROWS_PER_WAVE; ++r) {
    int id = idsBuf[wave * ROWS_PER_WAVE + r];
    const float4* src = (const float4*)(cb + (size_t)id * D_DIM);
    float4* dst = (float4*)(out_e + (size_t)(rowBase + r) * D_DIM);
#pragma unroll
    for (int j = 0; j < 4; ++j) dst[lane + 32 * j] = src[lane + 32 * j];
  }
}

extern "C" void kernel_launch(void* const* d_in, const int* in_sizes, int n_in,
                              void* d_out, int out_size, void* d_ws, size_t ws_size,
                              hipStream_t stream) {
  const float* x  = (const float*)d_in[0];
  const float* cb = (const float*)d_in[1];
  const int D = D_DIM;
  const int B = in_sizes[0] / D;   // 32768
  const int K = in_sizes[1] / D;   // 8192

  float* out_e   = (float*)d_out;
  float* out_ids = out_e + (size_t)B * D;

  float* cbsq = (float*)d_ws;                                        // K floats
  unsigned short* cbB = (unsigned short*)((char*)d_ws + (size_t)K * 4);  // K*512 bf16
  const size_t need = (size_t)K * 4 + (size_t)K * D * 2;
  const bool pre = ws_size >= need;

  vq_prep_kernel<<<(K + WAVES_PER_BLOCK - 1) / WAVES_PER_BLOCK, 256, 0, stream>>>(
      cb, cbsq, pre ? cbB : nullptr, K);
  if (pre)
    vq_main_kernel<true><<<B / ROWS_PER_BLOCK, 256, 0, stream>>>(x, cb, cbB, cbsq,
                                                                 out_e, out_ids, K);
  else
    vq_main_kernel<false><<<B / ROWS_PER_BLOCK, 256, 0, stream>>>(x, cb, nullptr, cbsq,
                                                                  out_e, out_ids, K);
}
